// SpectralGatingNetwork_51926154608658
// MI455X (gfx1250) — matrix operand order repaired and checked by reading, not compiled, not run on hardware
//
#include <hip/hip_runtime.h>
#include <math.h>

// ---------------------------------------------------------------------------
// SpectralGatingNetwork on MI455X (gfx1250, wave32, WMMA + async-LDS feed).
//
// Math transformation: rfft -> gate -> irfft along features is circular
// convolution with gt = irfft(gate), i.e. ht = h @ C, C[m][n] = gt[(n-m)%2048].
// concat([ht,ht]) @ W_fc1 = ht @ (W_fc1[:2048] + W_fc1[2048:]).
// Pipeline (all f16 inputs, f32 WMMA accumulation):
//   h   = x @ W_in + b_in                      (GEMM, bias epilogue)
//   ht  = h @ C                                (GEMM)
//   a1  = relu(ht @ W_fc1_sum)                 (GEMM, relu epilogue)
//   hty = ht * sigmoid(a1 @ W_fc2)             (GEMM, sigmoid*aux epilogue)
//   out = hty @ W_out + b_out                  (GEMM, bias, f32 out)
//
// GEMM feed uses GLOBAL_LOAD_ASYNC_TO_LDS_B128 (ASYNCcnt) into double-buffered
// LDS; WMMA fragment loads ride DScnt, so the two streams overlap.
// ---------------------------------------------------------------------------

typedef __attribute__((ext_vector_type(16))) _Float16 v16h;
typedef __attribute__((ext_vector_type(8)))  float    v8f;

#define DIM_   2048
#define BATCH_ 8192
#define REDC_  128          // DIM/16
#define PITCH  40           // LDS row pitch in halves (80B) to spread banks
#define TK     32           // K per WMMA step

enum { EPI_BIAS_H = 0, EPI_PLAIN_H = 1, EPI_RELU_H = 2, EPI_SIGMUL_H = 3, EPI_BIAS_F = 4 };

union FragU { uint4 q[2]; v16h v; };

// A fragment (16x32 f16, ISA 7.12.2): lane half kh holds K {kh*8..kh*8+7} and
// {16+kh*8..16+kh*8+7} for row (lane&15).
__device__ __forceinline__ v16h lds_a_frag(const _Float16* t, int row, int kh) {
  FragU f;
  f.q[0] = *(const uint4*)(t + row * PITCH + kh * 8);
  f.q[1] = *(const uint4*)(t + row * PITCH + kh * 8 + 16);
  return f.v;
}
// B fragment (32x16 f16): lane half kh holds K {kh*8..kh*8+7} and
// {16+kh*8..16+kh*8+7} for column (lane&15). LDS tile stores B^T rows (N-major).
__device__ __forceinline__ v16h lds_b_frag(const _Float16* t, int row, int kh) {
  FragU f;
  f.q[0] = *(const uint4*)(t + row * PITCH + kh * 8);
  f.q[1] = *(const uint4*)(t + row * PITCH + kh * 8 + 16);
  return f.v;
}

// Async memory->LDS b128 (GVS mode: saddr64 + vaddr32 offset). Tracked by
// ASYNCcnt; no VGPR data staging.
__device__ __forceinline__ void async_b128(unsigned lds_addr, unsigned voff,
                                           unsigned long long sbase) {
  asm volatile("global_load_async_to_lds_b128 %0, %1, %2"
               :: "v"(lds_addr), "v"(voff), "s"(sbase) : "memory");
}
__device__ __forceinline__ void wait_async0() {
  asm volatile("s_wait_asynccnt 0" ::: "memory");
}

template<int EPI>
__global__ __launch_bounds__(256)
void k_gemm(const _Float16* __restrict__ A,     // [M][K] row-major f16
            const _Float16* __restrict__ Bt,    // [N][K] row-major f16 (B^T)
            const float*    __restrict__ bias,  // [N] or nullptr
            const _Float16* __restrict__ aux,   // [M][N] f16 (sigmul) or nullptr
            _Float16*       __restrict__ outH,
            float*          __restrict__ outF,
            int M, int N, int K)
{
  __shared__ _Float16 ldsA[2][128 * PITCH];   // 10240 B per buffer
  __shared__ _Float16 ldsB[2][128 * PITCH];   // total LDS = 40960 B

  const int tid  = threadIdx.x;
  const int lane = tid & 31;
  const int wave = tid >> 5;      // 8 waves
  const int wm   = wave & 3;      // 4 wave-rows  -> 32 M-rows each
  const int wn   = wave >> 2;     // 2 wave-cols  -> 64 N-cols each
  const int bM   = blockIdx.y * 128;
  const int bN   = blockIdx.x * 128;

  v8f acc[2][4];
#pragma unroll
  for (int i = 0; i < 2; ++i)
#pragma unroll
    for (int j = 0; j < 4; ++j)
#pragma unroll
      for (int e = 0; e < 8; ++e) acc[i][j][e] = 0.0f;

  // Tile staging: 128x32 halves = 8KB per tile; 256 threads x 2 b128 chunks
  // per tile, issued as async-to-LDS loads.
  const int crow = tid >> 2;            // 0..63
  const int cseg = (tid & 3) * 8;       // half offset 0,8,16,24

  // Raw LDS byte offsets (low 32 bits of generic address; ISA: addr[31:0]).
  const unsigned ldsAbase = (unsigned)(size_t)&ldsA[0][0];
  const unsigned ldsBbase = (unsigned)(size_t)&ldsB[0][0];
  const unsigned BUFB = 128u * PITCH * 2u;                 // 10240 B per buffer
  const unsigned oL0 = (unsigned)((crow * PITCH + cseg) * 2);
  const unsigned oL1 = (unsigned)(((crow + 64) * PITCH + cseg) * 2);

  const unsigned long long Abase = (unsigned long long)A;
  const unsigned long long Bbase = (unsigned long long)Bt;
  const unsigned gA0 = (unsigned)(((size_t)(bM + crow)      * K + cseg) * 2);
  const unsigned gA1 = (unsigned)(((size_t)(bM + crow + 64) * K + cseg) * 2);
  const unsigned gB0 = (unsigned)(((size_t)(bN + crow)      * K + cseg) * 2);
  const unsigned gB1 = (unsigned)(((size_t)(bN + crow + 64) * K + cseg) * 2);

  // Preload K-step 0 into buffer 0.
  async_b128(ldsAbase + oL0, gA0, Abase);
  async_b128(ldsAbase + oL1, gA1, Abase);
  async_b128(ldsBbase + oL0, gB0, Bbase);
  async_b128(ldsBbase + oL1, gB1, Bbase);
  wait_async0();
  __syncthreads();

  const int nk   = K / TK;
  const int lrow = lane & 15;
  const int kh   = lane >> 4;

  for (int ks = 0; ks < nk; ++ks) {
    const int  cur  = ks & 1;
    const unsigned nxt = (unsigned)(cur ^ 1);
    const bool more = (ks + 1 < nk);

    if (more) {                 // async-feed tile ks+1 while computing tile ks
      const unsigned ko = (unsigned)((ks + 1) * TK * 2);
      async_b128(ldsAbase + nxt * BUFB + oL0, gA0 + ko, Abase);
      async_b128(ldsAbase + nxt * BUFB + oL1, gA1 + ko, Abase);
      async_b128(ldsBbase + nxt * BUFB + oL0, gB0 + ko, Bbase);
      async_b128(ldsBbase + nxt * BUFB + oL1, gB1 + ko, Bbase);
      if (ks + 2 < nk) {        // global_prefetch_b8 for tile after next
        __builtin_prefetch(A  + (size_t)(bM + crow) * K + cseg + (size_t)(ks + 2) * TK, 0, 1);
        __builtin_prefetch(Bt + (size_t)(bN + crow) * K + cseg + (size_t)(ks + 2) * TK, 0, 1);
      }
    }

    v16h af[2], bf[4];
    af[0] = lds_a_frag(ldsA[cur], wm * 32 + lrow,      kh);
    af[1] = lds_a_frag(ldsA[cur], wm * 32 + 16 + lrow, kh);
#pragma unroll
    for (int tn = 0; tn < 4; ++tn)
      bf[tn] = lds_b_frag(ldsB[cur], wn * 64 + tn * 16 + lrow, kh);

#pragma unroll
    for (int tm = 0; tm < 2; ++tm)
#pragma unroll
      for (int tn = 0; tn < 4; ++tn)
        acc[tm][tn] = __builtin_amdgcn_wmma_f32_16x16x32_f16(
            false, af[tm], false, bf[tn], (short)0, acc[tm][tn], false, false);

    wait_async0();              // own async loads into lds[nxt] landed
    __syncthreads();            // everyone done reading lds[cur] / writing nxt
  }

  // Epilogue. C/D layout: lanes 0-15 -> M = v, lanes 16-31 -> M = v + 8.
  const int r0 = bM + wm * 32 + kh * 8;
  const int c0 = bN + wn * 64 + lrow;
#pragma unroll
  for (int tm = 0; tm < 2; ++tm)
#pragma unroll
    for (int tn = 0; tn < 4; ++tn) {
      const int col = c0 + tn * 16;
      float bv = 0.0f;
      if (EPI == EPI_BIAS_H || EPI == EPI_BIAS_F) bv = bias[col];
#pragma unroll
      for (int v = 0; v < 8; ++v) {
        const int row = r0 + tm * 16 + v;
        const size_t o = (size_t)row * N + col;
        const float xv = acc[tm][tn][v];
        if (EPI == EPI_BIAS_H)        outH[o] = (_Float16)(xv + bv);
        else if (EPI == EPI_PLAIN_H)  outH[o] = (_Float16)xv;
        else if (EPI == EPI_RELU_H)   outH[o] = (_Float16)(xv > 0.0f ? xv : 0.0f);
        else if (EPI == EPI_SIGMUL_H) {
          const float y = 1.0f / (1.0f + __expf(-xv));
          outH[o] = (_Float16)((float)aux[o] * y);
        } else {
          outF[o] = xv + bv;
        }
      }
    }
}

// gt[n] = irfft(gate)[n] : direct inverse real-DFT, 2048 x 1025 (~2 MFLOP).
__global__ void k_time_kernel(const float* __restrict__ fg, float* __restrict__ gt) {
  const int n = blockIdx.x * blockDim.x + threadIdx.x;
  if (n >= DIM_) return;
  const float w = 6.2831853071795864769f * (float)n / (float)DIM_;
  float s = 0.0f;
  for (int f = 0; f <= DIM_ / 2; ++f) {
    float sn, cs;
    __sincosf(w * (float)f, &sn, &cs);
    const float m = (f == 0 || f == DIM_ / 2) ? 1.0f : 2.0f;
    s += m * (fg[2 * f] * cs - fg[2 * f + 1] * sn);
  }
  gt[n] = s / (float)DIM_;
}

// C^T[n][m] = gt[(n-m) & 2047] as f16 (B^T layout for the GEMM).
__global__ void k_circulantT(const float* __restrict__ gt, _Float16* __restrict__ BcT) {
  const int idx = blockIdx.x * blockDim.x + threadIdx.x;
  const int n = idx >> 11;
  const int m = idx & (DIM_ - 1);
  BcT[idx] = (_Float16)gt[(n - m) & (DIM_ - 1)];
}

__global__ void k_cvt_h(const float* __restrict__ in, _Float16* __restrict__ out, int n) {
  const int i = blockIdx.x * blockDim.x + threadIdx.x;
  if (i < n) out[i] = (_Float16)in[i];
}

// outT[c][r] = in[r][c] with f32->f16 conversion.
__global__ void k_transpose_h(const float* __restrict__ in, _Float16* __restrict__ outT,
                              int R, int C) {
  const int i = blockIdx.x * blockDim.x + threadIdx.x;
  if (i >= R * C) return;
  const int r = i / C, c = i - r * C;
  outT[(size_t)c * R + r] = (_Float16)in[i];
}

// W_fc1 fold (top+bottom halves) + transpose: outT[n][k] = W[k][n] + W[k+2048][n].
__global__ void k_fold_fc1T(const float* __restrict__ W, _Float16* __restrict__ outT) {
  const int i = blockIdx.x * blockDim.x + threadIdx.x;
  if (i >= REDC_ * DIM_) return;
  const int n = i / DIM_;
  const int k = i - n * DIM_;
  outT[i] = (_Float16)(W[(size_t)k * REDC_ + n] + W[(size_t)(k + DIM_) * REDC_ + n]);
}

extern "C" void kernel_launch(void* const* d_in, const int* in_sizes, int n_in,
                              void* d_out, int out_size, void* d_ws, size_t ws_size,
                              hipStream_t stream) {
  const float* x     = (const float*)d_in[0];
  const float* W_in  = (const float*)d_in[1];
  const float* b_in  = (const float*)d_in[2];
  const float* W_out = (const float*)d_in[3];
  const float* b_out = (const float*)d_in[4];
  const float* fgate = (const float*)d_in[5];
  const float* W_fc1 = (const float*)d_in[6];
  const float* W_fc2 = (const float*)d_in[7];
  float* out = (float*)d_out;

  char* ws = (char*)d_ws;
  size_t off = 0;
  auto take = [&](size_t bytes) -> void* {
    void* p = (void*)(ws + off);
    off += (bytes + 255) & ~(size_t)255;
    return p;
  };

  float*    gt     = (float*)   take((size_t)DIM_ * sizeof(float));
  _Float16* xh     = (_Float16*)take((size_t)BATCH_ * DIM_ * 2);
  _Float16* WinT   = (_Float16*)take((size_t)DIM_ * DIM_ * 2);
  _Float16* BcT    = (_Float16*)take((size_t)DIM_ * DIM_ * 2);
  _Float16* WoutT  = (_Float16*)take((size_t)DIM_ * DIM_ * 2);
  _Float16* Wfc1sT = (_Float16*)take((size_t)REDC_ * DIM_ * 2);
  _Float16* Wfc2T  = (_Float16*)take((size_t)DIM_ * REDC_ * 2);
  _Float16* hh     = (_Float16*)take((size_t)BATCH_ * DIM_ * 2);
  _Float16* hth    = (_Float16*)take((size_t)BATCH_ * DIM_ * 2);
  _Float16* a1h    = (_Float16*)take((size_t)BATCH_ * REDC_ * 2);
  _Float16* htyh   = (_Float16*)take((size_t)BATCH_ * DIM_ * 2);
  (void)ws_size; (void)in_sizes; (void)n_in; (void)out_size;

  // --- preprocessing: gate -> time kernel -> circulant; f32 -> f16 ----------
  k_time_kernel<<<DIM_ / 256, 256, 0, stream>>>(fgate, gt);
  k_circulantT<<<(DIM_ * DIM_) / 256, 256, 0, stream>>>(gt, BcT);
  k_cvt_h<<<(BATCH_ * DIM_) / 256, 256, 0, stream>>>(x, xh, BATCH_ * DIM_);
  k_transpose_h<<<(DIM_ * DIM_) / 256, 256, 0, stream>>>(W_in, WinT, DIM_, DIM_);
  k_transpose_h<<<(DIM_ * DIM_) / 256, 256, 0, stream>>>(W_out, WoutT, DIM_, DIM_);
  k_transpose_h<<<(REDC_ * DIM_) / 256, 256, 0, stream>>>(W_fc2, Wfc2T, REDC_, DIM_);
  k_fold_fc1T<<<(REDC_ * DIM_) / 256, 256, 0, stream>>>(W_fc1, Wfc1sT);

  // --- GEMM chain -----------------------------------------------------------
  const dim3 blk(256);
  const dim3 g_full(DIM_ / 128, BATCH_ / 128);   // (16, 64)
  const dim3 g_fc1(1, BATCH_ / 128);             // N = 128

  // h = x @ W_in + b_in
  k_gemm<EPI_BIAS_H><<<g_full, blk, 0, stream>>>(xh, WinT, b_in, nullptr,
                                                 hh, nullptr, BATCH_, DIM_, DIM_);
  // ht = h @ C  (spectral gate as circulant GEMM)
  k_gemm<EPI_PLAIN_H><<<g_full, blk, 0, stream>>>(hh, BcT, nullptr, nullptr,
                                                  hth, nullptr, BATCH_, DIM_, DIM_);
  // a1 = relu(ht @ W_fc1_sum)
  k_gemm<EPI_RELU_H><<<g_fc1, blk, 0, stream>>>(hth, Wfc1sT, nullptr, nullptr,
                                                a1h, nullptr, BATCH_, REDC_, DIM_);
  // hty = ht * sigmoid(a1 @ W_fc2)
  k_gemm<EPI_SIGMUL_H><<<g_full, blk, 0, stream>>>(a1h, Wfc2T, nullptr, hth,
                                                   htyh, nullptr, BATCH_, DIM_, REDC_);
  // out = hty @ W_out + b_out
  k_gemm<EPI_BIAS_F><<<g_full, blk, 0, stream>>>(htyh, WoutT, b_out, nullptr,
                                                 nullptr, out, BATCH_, DIM_, DIM_);
}
